// GINConv_4629974745742
// MI455X (gfx1250) — matrix operand order repaired and checked by reading, not compiled, not run on hardware
//
#include <hip/hip_runtime.h>
#include <hip/hip_bf16.h>

typedef __attribute__((ext_vector_type(16))) _Float16 v16h;
typedef __attribute__((ext_vector_type(8)))  float    v8f;
typedef float f32x4u __attribute__((ext_vector_type(4), aligned(4)));

union V16HU { uint4 q[2]; v16h v; };
union H8U   { uint4 q; _Float16 h[8]; };

#define IN_CH  50
#define OUT_CH 121
#define KPAD1  64      // 50 (+1 bias col) -> 64
#define NPAD   128     // 121 (+1 bias col) -> 128
#define WAVES  2       // waves per block
#define MROWS  32      // edges per wave supertile (2 x 16-col B tiles)

// ---------------------------------------------------------------------------
// Prep: transpose + pad + f16-convert weights, folding biases in as an extra
// K-row so the GEMMs produce  W^T @ [h;1]  =  W^T h + b  for free.
//   w1t[n][k] = k<50  ? W1[k][n] : (k==50  ? b1[n] : 0)   (128 x 64)
//   w2t[n][k] = k<121 ? W2[k][n] : (k==121 ? b2[n] : 0)   (128 x 128)
// Rows n >= 121 are all-zero.
// ---------------------------------------------------------------------------
__global__ void prep_weights(const float* __restrict__ W1,
                             const float* __restrict__ b1,
                             const float* __restrict__ W2,
                             const float* __restrict__ b2,
                             _Float16* __restrict__ w1t,
                             _Float16* __restrict__ w2t) {
  int i = blockIdx.x * blockDim.x + threadIdx.x;
  if (i < NPAD * KPAD1) {
    int n = i >> 6, k = i & 63;
    float v = 0.0f;
    if (n < OUT_CH) {
      if (k < IN_CH)        v = W1[k * OUT_CH + n];
      else if (k == IN_CH)  v = b1[n];
    }
    w1t[i] = (_Float16)v;
  }
  if (i < NPAD * NPAD) {
    int n = i >> 7, k = i & 127;
    float v = 0.0f;
    if (n < OUT_CH) {
      if (k < OUT_CH)       v = W2[k * OUT_CH + n];
      else if (k == OUT_CH) v = b2[n];
    }
    w2t[i] = (_Float16)v;
  }
}

// ---------------------------------------------------------------------------
// Main fused kernel: per wave, one 32-edge supertile, computed transposed:
//   D1^T = W1t(A) x h1^T(B)  -> relu -> h2 (row-major, b128 LDS stores)
//   D2^T = W2t(A) x h2^T(B)  -> vectorized global stores (channels contiguous
//                               per lane; bias already folded into the GEMM).
// ---------------------------------------------------------------------------
__global__ void __launch_bounds__(WAVES * 32)
gin_fused(const float* __restrict__ x,
          const long long* __restrict__ erow,
          const long long* __restrict__ ecol,
          const float* __restrict__ epsp,
          const _Float16* __restrict__ w1t_g,
          const _Float16* __restrict__ w2t_g,
          float* __restrict__ out,
          int E, int num_super) {
  extern __shared__ _Float16 smem[];
  _Float16* sW1    = smem;                               // 128*64  = 16KB
  _Float16* sW2    = smem + NPAD * KPAD1;                // 128*128 = 32KB
  _Float16* sStage = smem + NPAD * KPAD1 + NPAD * NPAD;  // 2 * 32*128 = 16KB

  const int tid  = threadIdx.x;
  const int lane = tid & 31;
  const int wave = tid >> 5;

  // Cooperative load of both weight matrices into LDS (48KB / 64 threads).
  {
    const uint4* s1 = (const uint4*)w1t_g;  uint4* d1 = (uint4*)sW1;
    for (int i = tid; i < (NPAD * KPAD1 * 2) / 16; i += WAVES * 32) d1[i] = s1[i];
    const uint4* s2 = (const uint4*)w2t_g;  uint4* d2 = (uint4*)sW2;
    for (int i = tid; i < (NPAD * NPAD * 2) / 16; i += WAVES * 32) d2[i] = s2[i];
  }
  __syncthreads();

  const float scale = 1.0f + epsp[0];

  _Float16* stage = sStage + wave * (MROWS * NPAD);  // wave-private 8KB

  const int mA = lane & 15;   // A-row (weight n) / B-col (edge) index
  const int hi = lane >> 4;   // lane half-group

  const v8f zero = {0.f, 0.f, 0.f, 0.f, 0.f, 0.f, 0.f, 0.f};

  for (int sup = blockIdx.x * WAVES + wave; sup < num_super;
       sup += gridDim.x * WAVES) {
    const int e0 = sup * MROWS;

    // ---- Stage 1: h1 = [(1+eps)*x[row] + x[col], 1], f16 -> stage[32][64] --
    // One lane per edge row; col 50 = 1.0 (bias column), cols 51..63 = 0.
    {
      int ee = e0 + lane; if (ee >= E) ee = E - 1;
      const float* xr = x + (size_t)erow[ee] * IN_CH;
      const float* xc = x + (size_t)ecol[ee] * IN_CH;
#pragma unroll
      for (int g = 0; g < 8; ++g) {
        H8U pack;
#pragma unroll
        for (int j = 0; j < 4; ++j) {
          const int cc = g * 8 + 2 * j;
          float v0 = 0.0f, v1 = 0.0f;
          if (cc < IN_CH) {            // 8B-aligned pair loads (row stride 200B)
            float2 a  = *(const float2*)(xr + cc);
            float2 bb = *(const float2*)(xc + cc);
            v0 = fmaf(scale, a.x, bb.x);
            if (cc + 1 < IN_CH) v1 = fmaf(scale, a.y, bb.y);
          } else if (cc == IN_CH) {
            v0 = 1.0f;                 // bias column
          }
          pack.h[2 * j]     = (_Float16)v0;
          pack.h[2 * j + 1] = (_Float16)v1;
        }
        *(uint4*)(stage + lane * KPAD1 + g * 8) = pack.q;  // ds_store_b128
      }
    }
    // Wave-private buffer; LDS ops are in-order within a wave -> no barrier.

    // ---- B1 = h1^T operands, register-resident (h1 LDS dead afterwards) ----
    V16HU B1[2][2];
#pragma unroll
    for (int mt = 0; mt < 2; ++mt)
#pragma unroll
      for (int ks = 0; ks < 2; ++ks) {
        const _Float16* bp = stage + (mt * 16 + mA) * KPAD1 + ks * 32 + hi * 8;
        B1[mt][ks].q[0] = *(const uint4*)(bp);
        B1[mt][ks].q[1] = *(const uint4*)(bp + 16);
      }

    // ---- GEMM1^T per n-tile: 4 WMMAs -> relu -> b128 store into h2 --------
#pragma unroll
    for (int t = 0; t < 8; ++t) {
      v8f acc[2] = {zero, zero};
#pragma unroll
      for (int ks = 0; ks < 2; ++ks) {
        const _Float16* ap = sW1 + (t * 16 + mA) * KPAD1 + ks * 32 + hi * 8;
        V16HU A;
        A.q[0] = *(const uint4*)(ap);
        A.q[1] = *(const uint4*)(ap + 16);
#pragma unroll
        for (int mt = 0; mt < 2; ++mt)   // A (weights) reused for both m tiles
          acc[mt] = __builtin_amdgcn_wmma_f32_16x16x32_f16(
              false, A.v, false, B1[mt][ks].v, (short)0, acc[mt], false, false);
      }
      // lane holds channels n = t*16 + hi*8 + r of edge row mt*16 + mA
#pragma unroll
      for (int mt = 0; mt < 2; ++mt) {
        H8U pack;
#pragma unroll
        for (int r = 0; r < 8; ++r) {
          float v = acc[mt][r];
          v = v > 0.0f ? v : 0.0f;
          pack.h[r] = (_Float16)v;
        }
        if (t == 7)                       // h2 col 121 = 1.0 (bias column)
          pack.h[1] = (hi == 1) ? (_Float16)1.0f : pack.h[1];
        *(uint4*)(stage + (mt * 16 + mA) * NPAD + t * 16 + hi * 8) = pack.q;
      }
    }

    // ---- B2 = h2^T operands (all h2 stores precede: DS in-order per wave) --
    V16HU B2[2][4];
#pragma unroll
    for (int mt = 0; mt < 2; ++mt)
#pragma unroll
      for (int ks = 0; ks < 4; ++ks) {
        const _Float16* bp = stage + (mt * 16 + mA) * NPAD + ks * 32 + hi * 8;
        B2[mt][ks].q[0] = *(const uint4*)(bp);
        B2[mt][ks].q[1] = *(const uint4*)(bp + 16);
      }

    // ---- GEMM2^T per n-tile: 8 WMMAs -> vectorized global store -----------
#pragma unroll
    for (int t = 0; t < 8; ++t) {
      v8f acc[2] = {zero, zero};
#pragma unroll
      for (int ks = 0; ks < 4; ++ks) {
        const _Float16* ap = sW2 + (t * 16 + mA) * NPAD + ks * 32 + hi * 8;
        V16HU A;
        A.q[0] = *(const uint4*)(ap);
        A.q[1] = *(const uint4*)(ap + 16);
#pragma unroll
        for (int mt = 0; mt < 2; ++mt)   // A (weights) reused for both m tiles
          acc[mt] = __builtin_amdgcn_wmma_f32_16x16x32_f16(
              false, A.v, false, B2[mt][ks].v, (short)0, acc[mt], false, false);
      }
      // lane holds 8 consecutive channels n = t*16 + hi*8 + r of one edge row
#pragma unroll
      for (int mt = 0; mt < 2; ++mt) {
        const int e = e0 + mt * 16 + mA;
        if (e < E) {
          float* op = out + (size_t)e * OUT_CH + t * 16 + hi * 8;
          if (t < 7) {                    // all 8 channels valid
            f32x4u lo = {acc[mt][0], acc[mt][1], acc[mt][2], acc[mt][3]};
            f32x4u ho = {acc[mt][4], acc[mt][5], acc[mt][6], acc[mt][7]};
            *(f32x4u*)(op)     = lo;      // 4B-aligned b128 stores
            *(f32x4u*)(op + 4) = ho;
          } else {                        // tail tile: n = 112..127, keep <121
#pragma unroll
            for (int r = 0; r < 8; ++r) {
              const int n = 112 + hi * 8 + r;
              if (n < OUT_CH) op[r] = acc[mt][r];
            }
          }
        }
      }
    }
  }
}

// ---------------------------------------------------------------------------
extern "C" void kernel_launch(void* const* d_in, const int* in_sizes, int n_in,
                              void* d_out, int out_size, void* d_ws, size_t ws_size,
                              hipStream_t stream) {
  (void)n_in; (void)out_size; (void)ws_size;
  const float*     x    = (const float*)d_in[0];
  const long long* eidx = (const long long*)d_in[1];   // int64 [2][E]
  const float*     W1   = (const float*)d_in[2];
  const float*     b1   = (const float*)d_in[3];
  const float*     W2   = (const float*)d_in[4];
  const float*     b2   = (const float*)d_in[5];
  const float*     eps  = (const float*)d_in[6];

  const int E = in_sizes[1] / 2;
  const long long* erow = eidx;
  const long long* ecol = eidx + E;

  _Float16* w1t = (_Float16*)d_ws;                 // 128*64 f16 = 16KB
  _Float16* w2t = w1t + NPAD * KPAD1;              // 128*128 f16 = 32KB

  prep_weights<<<(NPAD * NPAD + 255) / 256, 256, 0, stream>>>(
      W1, b1, W2, b2, w1t, w2t);

  const int num_super = (E + MROWS - 1) / MROWS;
  int grid = (num_super + WAVES - 1) / WAVES;
  if (grid > 4096) grid = 4096;
  const size_t shmem =
      (size_t)(NPAD * KPAD1 + NPAD * NPAD + WAVES * MROWS * NPAD) * sizeof(_Float16); // 64KB
  gin_fused<<<grid, WAVES * 32, shmem, stream>>>(
      x, erow, ecol, eps, w1t, w2t, (float*)d_out, E, num_super);
}
